// SparseMultiHeadAttn_52948356825281
// MI455X (gfx1250) — matrix operand order repaired and checked by reading, not compiled, not run on hardware
//
#include <hip/hip_runtime.h>
#include <hip/hip_bf16.h>

// Problem constants (must match reference)
#define S_ 4096
#define B_ 8
#define D_MODEL 1024
#define NHEAD 16
#define DH 64
#define CAP 256   // S_/NHEAD
#define HB 128    // NHEAD*B_

typedef __attribute__((ext_vector_type(16))) _Float16 v16h;
typedef __attribute__((ext_vector_type(8)))  float    v8f;

union Frag { uint4 q[2]; v16h h; };

// A-matrix 16x32 f16 fragment (wave32): lane holds row M=lane%16, K-half = lane/16.
// Elements 0..7  = halfs [k + 8*hl .. +7], elements 8..15 = halfs [k + 16 + 8*hl .. +7].
__device__ inline v16h load_frag_a(const uint4* row4, int k, int hl) {
  Frag f;
  f.q[0] = row4[(k >> 3) + hl];
  f.q[1] = row4[(k >> 3) + 2 + hl];
  return f.h;
}
// B-matrix 32x16 f16 fragment: lane holds column N=lane%16 (stored as a contiguous
// transposed row), elements 0..7 = halfs [k + 8*hl .. +7], 8..15 = [k + 16 + 8*hl .. +7].
__device__ inline v16h load_frag_b(const uint4* row4, int k, int hl) {
  Frag f;
  f.q[0] = row4[(k >> 3) + hl];
  f.q[1] = row4[(k >> 3) + 2 + hl];
  return f.h;
}

__device__ inline v8f wmma32(v16h a, v16h b, v8f c) {
  return __builtin_amdgcn_wmma_f32_16x16x32_f16(false, a, false, b, (short)0, c, false, false);
}

__device__ inline v8f vzero8() {
  v8f z = {0.f, 0.f, 0.f, 0.f, 0.f, 0.f, 0.f, 0.f};
  return z;
}

// ---------------- utility kernels ----------------

__global__ void k_zero(float* p, long n) {
  long i = (long)blockIdx.x * blockDim.x + threadIdx.x;
  long st = (long)gridDim.x * blockDim.x;
  for (; i < n; i += st) p[i] = 0.f;
}

__global__ void k_cvt(const float* __restrict__ s, _Float16* __restrict__ d, long n) {
  long i = (long)blockIdx.x * blockDim.x + threadIdx.x;
  long st = (long)gridDim.x * blockDim.x;
  for (; i < n; i += st) d[i] = (_Float16)s[i];
}

// Wq [H][D][64] -> Wq_t [H][64][D]
__global__ void k_cvt_wq(const float* __restrict__ w, _Float16* __restrict__ wt) {
  int gid = blockIdx.x * 256 + threadIdx.x;     // H*D*64 = 1048576
  int h = gid >> 16;
  int r = gid & 65535;
  int d = r >> 6, e = r & 63;
  wt[((h * 64 + e) << 10) + d] = (_Float16)w[gid];
}
// Wkv [H][D][128] -> Wkv_t [H][128][D]
__global__ void k_cvt_wkv(const float* __restrict__ w, _Float16* __restrict__ wt) {
  int gid = blockIdx.x * 256 + threadIdx.x;     // H*D*128 = 2097152
  int h = gid >> 17;
  int r = gid & 131071;
  int d = r >> 7, e = r & 127;
  wt[((h * 128 + e) << 10) + d] = (_Float16)w[gid];
}
// Wo [H][64][D] -> Wo_t [H][D][64]
__global__ void k_cvt_wo(const float* __restrict__ w, _Float16* __restrict__ wt) {
  int gid = blockIdx.x * 256 + threadIdx.x;     // H*64*D = 1048576
  int h = gid >> 16;
  int r = gid & 65535;
  int d = r >> 10, e = r & 1023;
  wt[(((long)h << 10) + e) * 64 + d] = (_Float16)w[gid];
}

// ---------------- gate: scores[b][h][s] = softmax_h(x[s,b,:] @ Wg) ----------------

__global__ __launch_bounds__(256) void k_gate(const float* __restrict__ x,
                                              const float* __restrict__ Wg,
                                              float* __restrict__ scores) {
  __shared__ _Float16 wgl[D_MODEL * NHEAD];   // 32 KB
  for (int i = threadIdx.x; i < D_MODEL * NHEAD; i += 256)
    wgl[i] = (_Float16)Wg[i];
  __syncthreads();
  int wave = threadIdx.x >> 5, lane = threadIdx.x & 31;
  int tok = blockIdx.x * 8 + wave;            // 0 .. B_*S_-1
  int b = tok / S_, s = tok % S_;
  const float* xr = x + ((long)s * B_ + b) * D_MODEL;
  float acc[NHEAD];
#pragma unroll
  for (int h = 0; h < NHEAD; h++) acc[h] = 0.f;
  for (int d = lane; d < D_MODEL; d += 32) {
    float xv = xr[d];
#pragma unroll
    for (int h = 0; h < NHEAD; h++) acc[h] += xv * (float)wgl[d * NHEAD + h];
  }
#pragma unroll
  for (int h = 0; h < NHEAD; h++)
    for (int off = 16; off > 0; off >>= 1) acc[h] += __shfl_xor(acc[h], off, 32);
  if (lane == 0) {
    float mx = acc[0];
#pragma unroll
    for (int h = 1; h < NHEAD; h++) mx = fmaxf(mx, acc[h]);
    float e[NHEAD], sum = 0.f;
#pragma unroll
    for (int h = 0; h < NHEAD; h++) { e[h] = __expf(acc[h] - mx); sum += e[h]; }
    float inv = 1.f / sum;
#pragma unroll
    for (int h = 0; h < NHEAD; h++)
      scores[((long)b * NHEAD + h) * S_ + s] = e[h] * inv;
  }
}

// ---------------- top-k (k=256 of 4096) per (b,h), indices ascending ----------------

__global__ __launch_bounds__(256) void k_topk(const float* __restrict__ scores,
                                              int* __restrict__ idx) {
  __shared__ int cnt[256];
  int t = threadIdx.x;
  int bh = blockIdx.x;                         // b*NHEAD + h
  const float* sc = scores + (long)bh * S_;
  int* out = idx + bh * CAP;
  out[t] = t;                                  // safe prefill
  // bisection: invariant count(>lo) >= CAP, count(>hi) < CAP
  float lo = -1.f, hi = 2.f;
  for (int it = 0; it < 32; ++it) {
    float mid = 0.5f * (lo + hi);
    int c = 0;
    for (int s = t; s < S_; s += 256) c += (sc[s] > mid) ? 1 : 0;
    cnt[t] = c; __syncthreads();
    for (int off = 128; off > 0; off >>= 1) {
      if (t < off) cnt[t] += cnt[t + off];
      __syncthreads();
    }
    int total = cnt[0]; __syncthreads();
    if (total >= CAP) lo = mid; else hi = mid;
  }
  int s0 = t * 16;                             // contiguous chunk -> ordered compaction
  // pass 1: strictly above threshold
  int myc = 0;
  for (int k = 0; k < 16; k++) myc += (sc[s0 + k] > hi) ? 1 : 0;
  cnt[t] = myc; __syncthreads();
  for (int off = 1; off < 256; off <<= 1) {
    int v = (t >= off) ? cnt[t - off] : 0;
    __syncthreads(); cnt[t] += v; __syncthreads();
  }
  int pos = cnt[t] - myc;
  int n1 = cnt[255];
  __syncthreads();
  for (int k = 0; k < 16; k++)
    if (sc[s0 + k] > hi) { if (pos < CAP) out[pos] = s0 + k; pos++; }
  // pass 2: boundary values (lo, hi] fill remaining slots in index order
  int myc2 = 0;
  for (int k = 0; k < 16; k++) { float v = sc[s0 + k]; myc2 += (v > lo && v <= hi) ? 1 : 0; }
  cnt[t] = myc2; __syncthreads();
  for (int off = 1; off < 256; off <<= 1) {
    int v = (t >= off) ? cnt[t - off] : 0;
    __syncthreads(); cnt[t] += v; __syncthreads();
  }
  pos = n1 + cnt[t] - myc2;
  __syncthreads();
  for (int k = 0; k < 16; k++) {
    float v = sc[s0 + k];
    if (v > lo && v <= hi) { if (pos < CAP) out[pos] = s0 + k; pos++; }
  }
}

// ------------- gathered Q/K/V projections via WMMA: one block per (h,b) -------------
// q[hb][c][64], k[hb][c][64], v_t[hb][64][c]  (f16)
// All 12 B-fragments are loaded before the 12 WMMAs so the loads overlap and the
// matrix pipe can issue WMMAs back-to-back behind partial loadcnt waits.

__global__ __launch_bounds__(512) void k_qkv(const _Float16* __restrict__ xh,
                                             const int* __restrict__ idx,
                                             const _Float16* __restrict__ Wq_t,
                                             const _Float16* __restrict__ Wkv_t,
                                             const float* __restrict__ bq,
                                             const float* __restrict__ bkv,
                                             _Float16* __restrict__ qb,
                                             _Float16* __restrict__ kb,
                                             _Float16* __restrict__ vtb) {
  int hb = blockIdx.x;
  int h = hb >> 3, b = hb & 7;
  int wave = threadIdx.x >> 5, lane = threadIdx.x & 31;
  int hl = lane >> 4, lm = lane & 15;
  int row = wave * 16 + lm;                    // 0..255 query row in capacity
  int tok = idx[(b * NHEAD + h) * CAP + row];
  const uint4* arow = (const uint4*)(xh + ((long)tok * B_ + b) * D_MODEL);

  const uint4* brq[4];
#pragma unroll
  for (int nt = 0; nt < 4; nt++)
    brq[nt] = (const uint4*)(Wq_t + ((long)h * DH + nt * 16 + lm) * D_MODEL);
  const uint4* brkv[8];
#pragma unroll
  for (int nt = 0; nt < 8; nt++)
    brkv[nt] = (const uint4*)(Wkv_t + ((long)h * 2 * DH + nt * 16 + lm) * D_MODEL);

  v8f accq[4], acckv[8];
#pragma unroll
  for (int nt = 0; nt < 4; nt++) accq[nt] = vzero8();
#pragma unroll
  for (int nt = 0; nt < 8; nt++) acckv[nt] = vzero8();

  for (int k = 0; k < D_MODEL; k += 32) {
    __builtin_prefetch((const char*)arow + (k + 64) * 2, 0, 1);  // global_prefetch_b8
    v16h a = load_frag_a(arow, k, hl);
    v16h bfq[4], bfkv[8];
#pragma unroll
    for (int nt = 0; nt < 4; nt++) bfq[nt] = load_frag_b(brq[nt], k, hl);
#pragma unroll
    for (int nt = 0; nt < 8; nt++) bfkv[nt] = load_frag_b(brkv[nt], k, hl);
#pragma unroll
    for (int nt = 0; nt < 4; nt++) accq[nt] = wmma32(a, bfq[nt], accq[nt]);
#pragma unroll
    for (int nt = 0; nt < 8; nt++) acckv[nt] = wmma32(a, bfkv[nt], acckv[nt]);
  }
  long base = (long)hb * CAP;
#pragma unroll
  for (int nt = 0; nt < 4; nt++) {
    int col = nt * 16 + lm;
    float bias = bq[h * DH + col];
#pragma unroll
    for (int r = 0; r < 8; r++) {
      int ro = wave * 16 + r + 8 * hl;         // C/D layout: M = r + 8*(lane/16)
      qb[(base + ro) * DH + col] = (_Float16)(accq[nt][r] + bias);
    }
  }
#pragma unroll
  for (int nt = 0; nt < 8; nt++) {
    int n = nt * 16 + lm;
    float bias = bkv[h * 2 * DH + n];
#pragma unroll
    for (int r = 0; r < 8; r++) {
      int ro = wave * 16 + r + 8 * hl;
      float v = acckv[nt][r] + bias;
      if (n < DH) kb[(base + ro) * DH + n] = (_Float16)v;
      else        vtb[((long)hb * DH + (n - DH)) * CAP + ro] = (_Float16)v;  // transposed V
    }
  }
}

// ---------------- att[hb][i][j] = scale * q[i,:] . k[j,:] ----------------
// j-tile loop unrolled x4: 8 fragment loads in flight, 4 independent WMMA chains.

__global__ __launch_bounds__(512) void k_att(const _Float16* __restrict__ qb,
                                             const _Float16* __restrict__ kb,
                                             float* __restrict__ attb) {
  int hb = blockIdx.x;
  int wave = threadIdx.x >> 5, lane = threadIdx.x & 31;
  int hl = lane >> 4, lm = lane & 15;
  int i0 = wave * 16;
  const uint4* arow = (const uint4*)(qb + ((long)hb * CAP + i0 + lm) * DH);
  v16h a0 = load_frag_a(arow, 0, hl);
  v16h a1 = load_frag_a(arow, 32, hl);
  const float scale = 0.125f;                  // 1/sqrt(64)
  for (int jt = 0; jt < 16; jt += 4) {
    v16h b0[4], b1[4];
#pragma unroll
    for (int u = 0; u < 4; u++) {
      int j = (jt + u) * 16 + lm;
      const uint4* brow = (const uint4*)(kb + ((long)hb * CAP + j) * DH);
      b0[u] = load_frag_b(brow, 0, hl);
      b1[u] = load_frag_b(brow, 32, hl);
    }
    v8f acc[4];
#pragma unroll
    for (int u = 0; u < 4; u++) acc[u] = vzero8();
#pragma unroll
    for (int u = 0; u < 4; u++) acc[u] = wmma32(a0, b0[u], acc[u]);
#pragma unroll
    for (int u = 0; u < 4; u++) acc[u] = wmma32(a1, b1[u], acc[u]);
#pragma unroll
    for (int u = 0; u < 4; u++) {
      int j = (jt + u) * 16 + lm;
#pragma unroll
      for (int r = 0; r < 8; r++) {
        int i = i0 + r + 8 * hl;
        attb[((long)hb * CAP + i) * CAP + j] = acc[u][r] * scale;
      }
    }
  }
}

// ----- faithful to reference: softmax over the BATCH axis (axis=1 of [H,B,C,C]) -----

__global__ __launch_bounds__(256) void k_softmax_b(const float* __restrict__ attb,
                                                   _Float16* __restrict__ pb) {
  long gid = (long)blockIdx.x * 256 + threadIdx.x;   // H*C*C = 1048576
  int h = (int)(gid >> 16);
  int rem = (int)(gid & 65535);                      // i*256 + j
  float v[B_], mx = -1e30f;
#pragma unroll
  for (int b = 0; b < B_; b++) {
    v[b] = attb[(((long)h * B_ + b) << 16) + rem];
    mx = fmaxf(mx, v[b]);
  }
  float sum = 0.f;
#pragma unroll
  for (int b = 0; b < B_; b++) { v[b] = __expf(v[b] - mx); sum += v[b]; }
  float inv = 1.f / sum;
#pragma unroll
  for (int b = 0; b < B_; b++)
    pb[(((long)h * B_ + b) << 16) + rem] = (_Float16)(v[b] * inv);
}

// ---------------- av[hb][i][d] = p[hb][i][:] @ v[hb][:][d] ----------------

__global__ __launch_bounds__(512) void k_av(const _Float16* __restrict__ pb,
                                            const _Float16* __restrict__ vtb,
                                            _Float16* __restrict__ avb) {
  int hb = blockIdx.x;
  int wave = threadIdx.x >> 5, lane = threadIdx.x & 31;
  int hl = lane >> 4, lm = lane & 15;
  int i0 = wave * 16;
  const uint4* arow = (const uint4*)(pb + ((long)hb * CAP + i0 + lm) * CAP);
  const uint4* brow[4];
#pragma unroll
  for (int nt = 0; nt < 4; nt++)
    brow[nt] = (const uint4*)(vtb + ((long)hb * DH + nt * 16 + lm) * CAP);
  v8f acc[4];
#pragma unroll
  for (int nt = 0; nt < 4; nt++) acc[nt] = vzero8();
  for (int k = 0; k < CAP; k += 32) {
    v16h a = load_frag_a(arow, k, hl);
    v16h bf[4];
#pragma unroll
    for (int nt = 0; nt < 4; nt++) bf[nt] = load_frag_b(brow[nt], k, hl);
#pragma unroll
    for (int nt = 0; nt < 4; nt++) acc[nt] = wmma32(a, bf[nt], acc[nt]);
  }
#pragma unroll
  for (int nt = 0; nt < 4; nt++) {
    int d = nt * 16 + lm;
#pragma unroll
    for (int r = 0; r < 8; r++) {
      int i = i0 + r + 8 * hl;
      avb[((long)hb * CAP + i) * DH + d] = (_Float16)acc[nt][r];
    }
  }
}

// ---- ao = av @ Wo + bo, scatter-added at token slots (heads may collide -> atomics) ----
// Column-tile loop unrolled x2: two independent accumulator chains per step.

__global__ __launch_bounds__(512) void k_ao(const _Float16* __restrict__ avb,
                                            const _Float16* __restrict__ Wo_t,
                                            const float* __restrict__ bo,
                                            const int* __restrict__ idx,
                                            float* __restrict__ outs) {
  int hb = blockIdx.x;
  int h = hb >> 3, b = hb & 7;
  int wave = threadIdx.x >> 5, lane = threadIdx.x & 31;
  int hl = lane >> 4, lm = lane & 15;
  int i0 = wave * 16;
  const uint4* arow = (const uint4*)(avb + ((long)hb * CAP + i0 + lm) * DH);
  v16h a0 = load_frag_a(arow, 0, hl);
  v16h a1 = load_frag_a(arow, 32, hl);
  int toks[8];
#pragma unroll
  for (int r = 0; r < 8; r++)
    toks[r] = idx[(b * NHEAD + h) * CAP + i0 + r + 8 * hl];
  for (int nt = 0; nt < 64; nt += 2) {
    int e0 = nt * 16 + lm;
    int e1 = (nt + 1) * 16 + lm;
    const uint4* br0 = (const uint4*)(Wo_t + (((long)h << 10) + e0) * DH);
    const uint4* br1 = (const uint4*)(Wo_t + (((long)h << 10) + e1) * DH);
    v16h c00 = load_frag_b(br0, 0, hl);
    v16h c01 = load_frag_b(br0, 32, hl);
    v16h c10 = load_frag_b(br1, 0, hl);
    v16h c11 = load_frag_b(br1, 32, hl);
    v8f acc0 = vzero8(), acc1 = vzero8();
    acc0 = wmma32(a0, c00, acc0);
    acc1 = wmma32(a0, c10, acc1);
    acc0 = wmma32(a1, c01, acc0);
    acc1 = wmma32(a1, c11, acc1);
    float bias0 = bo[e0], bias1 = bo[e1];
#pragma unroll
    for (int r = 0; r < 8; r++) {
      long rowoff = ((long)b * S_ + toks[r]) * D_MODEL;
      atomicAdd(&outs[rowoff + e0], acc0[r] + bias0);
      atomicAdd(&outs[rowoff + e1], acc1[r] + bias1);
    }
  }
}

// ---------------- residual + LayerNorm, write out [S,B,D] ----------------

__global__ __launch_bounds__(256) void k_final(const float* __restrict__ x,
                                               const float* __restrict__ outs,
                                               const float* __restrict__ ln_g,
                                               const float* __restrict__ ln_b,
                                               float* __restrict__ out) {
  __shared__ float red[256];
  int tk = blockIdx.x;                          // b*S_ + s
  int t = threadIdx.x;
  int b = tk / S_, s = tk % S_;
  const float* xr = x + ((long)s * B_ + b) * D_MODEL;
  const float* orow = outs + ((long)b * S_ + s) * D_MODEL;
  float y[4]; float sum = 0.f;
#pragma unroll
  for (int j = 0; j < 4; j++) {
    int d = t + j * 256;
    y[j] = xr[d] + orow[d];
    sum += y[j];
  }
  red[t] = sum; __syncthreads();
  for (int off = 128; off > 0; off >>= 1) {
    if (t < off) red[t] += red[t + off];
    __syncthreads();
  }
  float mu = red[0] * (1.f / D_MODEL);
  __syncthreads();
  float sq = 0.f;
#pragma unroll
  for (int j = 0; j < 4; j++) { float dv = y[j] - mu; sq += dv * dv; }
  red[t] = sq; __syncthreads();
  for (int off = 128; off > 0; off >>= 1) {
    if (t < off) red[t] += red[t + off];
    __syncthreads();
  }
  float var = red[0] * (1.f / D_MODEL);
  float rs = rsqrtf(var + 1e-5f);
  float* wr = out + ((long)s * B_ + b) * D_MODEL;
#pragma unroll
  for (int j = 0; j < 4; j++) {
    int d = t + j * 256;
    wr[d] = (y[j] - mu) * rs * ln_g[d] + ln_b[d];
  }
}

// ---------------- host orchestration ----------------

extern "C" void kernel_launch(void* const* d_in, const int* in_sizes, int n_in,
                              void* d_out, int out_size, void* d_ws, size_t ws_size,
                              hipStream_t stream) {
  (void)in_sizes; (void)n_in; (void)out_size; (void)ws_size;
  const float* x    = (const float*)d_in[0];
  // d_in[1] = attn_mask (all False -> skipped, faithful to reference)
  const float* Wg   = (const float*)d_in[2];
  const float* Wq   = (const float*)d_in[3];
  const float* bq   = (const float*)d_in[4];
  const float* Wkv  = (const float*)d_in[5];
  const float* bkv  = (const float*)d_in[6];
  const float* Wo   = (const float*)d_in[7];
  const float* bo   = (const float*)d_in[8];
  const float* ln_g = (const float*)d_in[9];
  const float* ln_b = (const float*)d_in[10];
  float* out = (float*)d_out;

  char* p = (char*)d_ws;
  auto alloc = [&](size_t bytes) -> char* {
    char* r = p;
    p += (bytes + 255) & ~(size_t)255;
    return r;
  };
  _Float16* xh     = (_Float16*)alloc(2ul * S_ * B_ * D_MODEL);        // 67 MB
  _Float16* Wq_t   = (_Float16*)alloc(2ul * NHEAD * DH * D_MODEL);     // 2 MB
  _Float16* Wkv_t  = (_Float16*)alloc(2ul * NHEAD * 2 * DH * D_MODEL); // 4 MB
  _Float16* Wo_t   = (_Float16*)alloc(2ul * NHEAD * D_MODEL * DH);     // 2 MB
  float*    scores = (float*)   alloc(4ul * B_ * NHEAD * S_);          // 2 MB
  int*      idxb   = (int*)     alloc(4ul * B_ * NHEAD * CAP);         // 128 KB
  _Float16* qb     = (_Float16*)alloc(2ul * HB * CAP * DH);            // 4 MB
  _Float16* kb     = (_Float16*)alloc(2ul * HB * CAP * DH);            // 4 MB
  _Float16* vtb    = (_Float16*)alloc(2ul * HB * DH * CAP);            // 4 MB
  float*    attb   = (float*)   alloc(4ul * HB * CAP * CAP);           // 33.5 MB
  _Float16* pb     = (_Float16*)alloc(2ul * HB * CAP * CAP);           // 16.8 MB
  _Float16* avb    = (_Float16*)alloc(2ul * HB * CAP * DH);            // 4 MB
  float*    outs   = (float*)   alloc(4ul * B_ * S_ * D_MODEL);        // 134 MB

  k_zero   <<<2048, 256, 0, stream>>>(outs, (long)B_ * S_ * D_MODEL);
  k_cvt    <<<4096, 256, 0, stream>>>(x, xh, (long)S_ * B_ * D_MODEL);
  k_cvt_wq <<<4096, 256, 0, stream>>>(Wq, Wq_t);
  k_cvt_wkv<<<8192, 256, 0, stream>>>(Wkv, Wkv_t);
  k_cvt_wo <<<4096, 256, 0, stream>>>(Wo, Wo_t);
  k_gate   <<<(B_ * S_) / 8, 256, 0, stream>>>(x, Wg, scores);
  k_topk   <<<B_ * NHEAD, 256, 0, stream>>>(scores, idxb);
  k_qkv    <<<HB, 512, 0, stream>>>(xh, idxb, Wq_t, Wkv_t, bq, bkv, qb, kb, vtb);
  k_att    <<<HB, 512, 0, stream>>>(qb, kb, attb);
  k_softmax_b<<<(NHEAD * CAP * CAP) / 256, 256, 0, stream>>>(attb, pb);
  k_av     <<<HB, 512, 0, stream>>>(pb, vtb, avb);
  k_ao     <<<HB, 512, 0, stream>>>(avb, Wo_t, bo, idxb, outs);
  k_final  <<<B_ * S_, 256, 0, stream>>>(x, outs, ln_g, ln_b, out);
}
